// GraphConvolutionLayer_51393578664142
// MI455X (gfx1250) — matrix operand order repaired and checked by reading, not compiled, not run on hardware
//
#include <hip/hip_runtime.h>

typedef __attribute__((ext_vector_type(16))) _Float16 v16h;
typedef __attribute__((ext_vector_type(8)))  _Float16 v8h;
typedef __attribute__((ext_vector_type(4)))  _Float16 v4h;
typedef __attribute__((ext_vector_type(8)))  float    v8f;

#define B_   32
#define N_   512
#define FIN  256
#define FOUT 256
#define E_   2

#define WMMA_F16(a, b, c) __builtin_amdgcn_wmma_f32_16x16x32_f16( \
    false, (a), false, (b), (short)0, (c), false, false)

// ---------------- fp32 -> fp16 conversion (4 elems/thread) ----------------
__global__ __launch_bounds__(256) void cvt_f32_f16(const float* __restrict__ src,
                                                   _Float16* __restrict__ dst, int n4) {
    int i = blockIdx.x * 256 + threadIdx.x;
    if (i >= n4) return;
    float4 v = ((const float4*)src)[i];
    v4h o;
    o[0] = (_Float16)v.x; o[1] = (_Float16)v.y;
    o[2] = (_Float16)v.z; o[3] = (_Float16)v.w;
    ((v4h*)dst)[i] = o;
}

// ------------- fp32 -> fp16 convert + transpose: W[K][Nc] -> WT[Nc][K] ------
__global__ __launch_bounds__(256) void cvt_transpose_f16(const float* __restrict__ src,
                                                         _Float16* __restrict__ dst,
                                                         int K, int Nc, int total) {
    int i = blockIdx.x * 256 + threadIdx.x;
    if (i >= total) return;
    int k = i / Nc, n = i % Nc;
    dst[(size_t)n * K + k] = (_Float16)src[i];
}

// ---------------- WMMA fragment loaders (contiguous k = 16*khalf + s, same for A and B) ----------------
// A fragment: 16x32 f16 from row-major source. lane m = lane&15, khalf = lane>>4.
// Elements 0..15 = Arow[k0 + khalf*16 + 0..15] (contiguous, matches B fragment)
__device__ __forceinline__ v16h load_a_frag(const _Float16* __restrict__ Arow,
                                            int k0, int khalf) {
    return *(const v16h*)(Arow + k0 + (khalf << 4));
}

// B fragment: 32x16 f16 from TRANSPOSED source BT[N][K]. lane n = lane&15.
// VGPR v holds K = k0 + khalf*16 + 2v (+1): 16 contiguous halves -> one 32B load.
__device__ __forceinline__ v16h load_bt_frag(const _Float16* __restrict__ BTrow,
                                             int k0, int khalf) {
    return *(const v16h*)(BTrow + k0 + (khalf << 4));
}

// ---- software-pipelined K-loop: 64x32 tile, 8 accumulators (4 M x 2 N) -----
// 12 b128-class loads feed 8 v_wmma per 32-K step (1.5 loads/wmma); next-step
// fragments are fetched before current-step WMMAs issue.
template <int K>
__device__ __forceinline__ void gemm_seg2(const _Float16* __restrict__ a0,
                                          const _Float16* __restrict__ a1,
                                          const _Float16* __restrict__ a2,
                                          const _Float16* __restrict__ a3,
                                          const _Float16* __restrict__ bt0,
                                          const _Float16* __restrict__ bt1,
                                          int khalf, v8f (&acc)[8]) {
    v16h b0 = load_bt_frag(bt0, 0, khalf);
    v16h b1 = load_bt_frag(bt1, 0, khalf);
    v16h f0 = load_a_frag(a0, 0, khalf);
    v16h f1 = load_a_frag(a1, 0, khalf);
    v16h f2 = load_a_frag(a2, 0, khalf);
    v16h f3 = load_a_frag(a3, 0, khalf);
    for (int k0 = 32; k0 < K; k0 += 32) {
        v16h c0 = load_bt_frag(bt0, k0, khalf);
        v16h c1 = load_bt_frag(bt1, k0, khalf);
        v16h g0 = load_a_frag(a0, k0, khalf);
        v16h g1 = load_a_frag(a1, k0, khalf);
        v16h g2 = load_a_frag(a2, k0, khalf);
        v16h g3 = load_a_frag(a3, k0, khalf);
        acc[0] = WMMA_F16(f0, b0, acc[0]);
        acc[1] = WMMA_F16(f1, b0, acc[1]);
        acc[2] = WMMA_F16(f2, b0, acc[2]);
        acc[3] = WMMA_F16(f3, b0, acc[3]);
        acc[4] = WMMA_F16(f0, b1, acc[4]);
        acc[5] = WMMA_F16(f1, b1, acc[5]);
        acc[6] = WMMA_F16(f2, b1, acc[6]);
        acc[7] = WMMA_F16(f3, b1, acc[7]);
        b0 = c0; b1 = c1; f0 = g0; f1 = g1; f2 = g2; f3 = g3;
    }
    acc[0] = WMMA_F16(f0, b0, acc[0]);
    acc[1] = WMMA_F16(f1, b0, acc[1]);
    acc[2] = WMMA_F16(f2, b0, acc[2]);
    acc[3] = WMMA_F16(f3, b0, acc[3]);
    acc[4] = WMMA_F16(f0, b1, acc[4]);
    acc[5] = WMMA_F16(f1, b1, acc[5]);
    acc[6] = WMMA_F16(f2, b1, acc[6]);
    acc[7] = WMMA_F16(f3, b1, acc[7]);
}

// ---------------- GEMM1: hT[b,e] = (n16[b] @ W_adj[e] + b_adj[e])^T ---------
// Each wave: 64x32 output tile.
__global__ __launch_bounds__(256) void gemm_h_kernel(const _Float16* __restrict__ n16,
                                                     const _Float16* __restrict__ WadjT16,
                                                     const float*    __restrict__ b_adj,
                                                     _Float16*       __restrict__ hT16) {
    const int lane = threadIdx.x & 31;
    const int wave = blockIdx.x * 8 + (threadIdx.x >> 5);
    const int tiles_per_be = (N_ / 64) * (FOUT / 32);   // 8*8 = 64
    const int be = wave / tiles_per_be;
    if (be >= B_ * E_) return;
    const int t  = wave % tiles_per_be;
    const int mt = t / (FOUT / 32);   // 0..7
    const int nt = t % (FOUT / 32);   // 0..7
    const int b = be / E_, e = be % E_;

    const int m     = lane & 15;
    const int khalf = lane >> 4;
    const int nl    = lane & 15;

    const _Float16* A   = n16 + (size_t)b * N_ * FIN + (size_t)(mt * 64 + m) * FIN;
    const _Float16* BT  = WadjT16 + (size_t)e * FOUT * FIN;
    const _Float16* bt0 = BT + (size_t)(nt * 32 + nl) * FIN;
    const _Float16* bt1 = BT + (size_t)(nt * 32 + 16 + nl) * FIN;

    v8f acc[8] = {};
    gemm_seg2<FIN>(A, A + 16 * FIN, A + 32 * FIN, A + 48 * FIN, bt0, bt1, khalf, acc);

    // acc[j*4+s][r] = element (M = mt*64 + s*16 + r + 8*khalf, N = nt*32 + j*16 + nl).
    // Transposed store hT[nn][M]: r contiguous -> one packed 16B store per acc.
#pragma unroll
    for (int j = 0; j < 2; ++j) {
        const int nn = nt * 32 + j * 16 + nl;
        const float bias = b_adj[e * FOUT + nn];
        _Float16* Hrow = hT16 + ((size_t)(b * E_ + e) * FOUT + nn) * N_;
#pragma unroll
        for (int s = 0; s < 4; ++s) {
            v8h hv;
#pragma unroll
            for (int r = 0; r < 8; ++r) hv[r] = (_Float16)(acc[j * 4 + s][r] + bias);
            *(v8h*)(Hrow + mt * 64 + s * 16 + (khalf << 3)) = hv;
        }
    }
}

// ---- GEMM2 (fused): out[b] = relu( Σ_e adj16[b,e]@h[b,e] + n16[b]@W2 + b2 )
__global__ __launch_bounds__(256) void gemm_out_kernel(const _Float16* __restrict__ adj16,
                                                       const _Float16* __restrict__ hT16,
                                                       const _Float16* __restrict__ n16,
                                                       const _Float16* __restrict__ W2T16,
                                                       const float*    __restrict__ b2,
                                                       float*          __restrict__ out) {
    const int lane = threadIdx.x & 31;
    const int wave = blockIdx.x * 8 + (threadIdx.x >> 5);
    const int tiles_per_b = (N_ / 64) * (FOUT / 32);    // 64
    const int b = wave / tiles_per_b;
    if (b >= B_) return;
    const int t  = wave % tiles_per_b;
    const int mt = t / (FOUT / 32);
    const int nt = t % (FOUT / 32);

    const int m     = lane & 15;
    const int khalf = lane >> 4;
    const int nl    = lane & 15;

    v8f acc[8] = {};

    // edge-type aggregation: Σ_e adj[b,e] @ h[b,e]   (K = 512 each)
#pragma unroll
    for (int e = 0; e < E_; ++e) {
        const _Float16* A = adj16 + (size_t)(b * E_ + e) * N_ * N_
                                  + (size_t)(mt * 64 + m) * N_;
        const _Float16* BT  = hT16 + (size_t)(b * E_ + e) * FOUT * N_;
        const _Float16* bt0 = BT + (size_t)(nt * 32 + nl) * N_;
        const _Float16* bt1 = BT + (size_t)(nt * 32 + 16 + nl) * N_;
        gemm_seg2<N_>(A, A + 16 * N_, A + 32 * N_, A + 48 * N_, bt0, bt1, khalf, acc);
    }

    // fused second linear: n[b] @ W2   (K = 256)
    {
        const _Float16* A = n16 + (size_t)b * N_ * FIN + (size_t)(mt * 64 + m) * FIN;
        const _Float16* bt0 = W2T16 + (size_t)(nt * 32 + nl) * FIN;
        const _Float16* bt1 = W2T16 + (size_t)(nt * 32 + 16 + nl) * FIN;
        gemm_seg2<FIN>(A, A + 16 * FIN, A + 32 * FIN, A + 48 * FIN, bt0, bt1, khalf, acc);
    }

    // fp32 epilogue; lanes 0-15 cover 16 consecutive nn -> coalesced stores.
    float* O = out + (size_t)b * N_ * FOUT;
#pragma unroll
    for (int j = 0; j < 2; ++j) {
        const int nn = nt * 32 + j * 16 + nl;
        const float bias = b2[nn];
#pragma unroll
        for (int s = 0; s < 4; ++s)
#pragma unroll
            for (int r = 0; r < 8; ++r) {
                int mm = mt * 64 + s * 16 + r + (khalf << 3);
                float v = acc[j * 4 + s][r] + bias;
                O[(size_t)mm * FOUT + nn] = v > 0.0f ? v : 0.0f;
            }
    }
}

// ---------------------------------------------------------------------------
extern "C" void kernel_launch(void* const* d_in, const int* in_sizes, int n_in,
                              void* d_out, int out_size, void* d_ws, size_t ws_size,
                              hipStream_t stream) {
    const float* n_f32    = (const float*)d_in[0];   // [B,N,FIN]
    const float* adj_f32  = (const float*)d_in[1];   // [B,E,N,N]
    const float* Wadj_f32 = (const float*)d_in[2];   // [E,FIN,FOUT]
    const float* badj_f32 = (const float*)d_in[3];   // [E,FOUT]
    const float* W2_f32   = (const float*)d_in[4];   // [FIN,FOUT]
    const float* b2_f32   = (const float*)d_in[5];   // [FOUT]
    float* out = (float*)d_out;

    const size_t n_elems    = (size_t)B_ * N_ * FIN;          // 4,194,304
    const size_t adj_elems  = (size_t)B_ * E_ * N_ * N_;      // 16,777,216
    const size_t wadj_elems = (size_t)E_ * FIN * FOUT;        //   131,072
    const size_t w2_elems   = (size_t)FIN * FOUT;             //    65,536

    char* ws = (char*)d_ws;
    _Float16* n16     = (_Float16*)ws;                     ws += n_elems    * 2;
    _Float16* adj16   = (_Float16*)ws;                     ws += adj_elems  * 2;
    _Float16* WadjT16 = (_Float16*)ws;                     ws += wadj_elems * 2;
    _Float16* W2T16   = (_Float16*)ws;                     ws += w2_elems   * 2;
    _Float16* hT16    = (_Float16*)ws;

    // 1) convert A-side operands (row-major kept); adj traffic 67MB -> 33.5MB
    cvt_f32_f16<<<(int)(n_elems   / 4 / 256), 256, 0, stream>>>(n_f32,   n16,   (int)(n_elems   / 4));
    cvt_f32_f16<<<(int)(adj_elems / 4 / 256), 256, 0, stream>>>(adj_f32, adj16, (int)(adj_elems / 4));

    // 2) convert + transpose B-side weights so B fragments are contiguous 32B
    for (int e = 0; e < E_; ++e)
        cvt_transpose_f16<<<(int)((w2_elems + 255) / 256), 256, 0, stream>>>(
            Wadj_f32 + (size_t)e * FIN * FOUT, WadjT16 + (size_t)e * FOUT * FIN,
            FIN, FOUT, (int)w2_elems);
    cvt_transpose_f16<<<(int)((w2_elems + 255) / 256), 256, 0, stream>>>(
        W2_f32, W2T16, FIN, FOUT, (int)w2_elems);

    // 3) hT = (n @ W_adj + b_adj)^T : 4096 waves (64x32 tiles), 8 waves/block
    {
        int waves = B_ * E_ * (N_ / 64) * (FOUT / 32);  // 4096
        gemm_h_kernel<<<waves / 8, 256, 0, stream>>>(n16, WadjT16, badj_f32, hT16);
    }

    // 4) out = relu(Σ_e adj@h + n@W2 + b2) : fused, 2048 waves
    {
        int waves = B_ * (N_ / 64) * (FOUT / 32);       // 2048
        gemm_out_kernel<<<waves / 8, 256, 0, stream>>>(adj16, hT16, n16, W2T16, b2_f32, out);
    }
}
